// MoETransitionRewardModel_10170482557096
// MI455X (gfx1250) — matrix operand order repaired and checked by reading, not compiled, not run on hardware
//
#include <hip/hip_runtime.h>
#include <hip/hip_bf16.h>
#include <math.h>

// ---------------------------------------------------------------------------
// MoE transition/reward model forward on gfx1250 (MI455X).
// Compute-bound (~760 GFLOP, ~3000 FLOP/byte) -> all heavy GEMMs go through
// v_wmma_f32_16x16x32_bf16 with f32 accumulation.
// Weights are pre-transposed (n-major, zero-padded rows) at the per-launch
// bf16 conversion, so BOTH tiles stage into LDS as straight contiguous copies
// via GLOBAL_LOAD_ASYNC_TO_LDS_B128 (zero data VGPRs, ASYNCcnt-tracked),
// double-buffered: async(next) -> s_wait_asynccnt(8) -> barrier -> wmma(cur).
// ---------------------------------------------------------------------------

typedef __attribute__((ext_vector_type(16))) __bf16 v16bf;
typedef __attribute__((ext_vector_type(8)))  float  v8f;
typedef __attribute__((ext_vector_type(4)))  int    v4i;

#define TM 128
#define TN 128
#define TK 64
#define KPAD 8
#define LDT (TK + KPAD) /* 72 bf16 = 144B, multiple of 16B: b128-friendly */

#if defined(__has_builtin)
#if __has_builtin(__builtin_amdgcn_global_load_async_to_lds_b128)
#define USE_ASYNC_LDS 1
#endif
#endif

#define AS1 __attribute__((address_space(1)))
#define AS3 __attribute__((address_space(3)))

__device__ __forceinline__ void wait_async_le(int tag) {
#if defined(USE_ASYNC_LDS)
#if __has_builtin(__builtin_amdgcn_s_wait_asynccnt)
  if (tag) __builtin_amdgcn_s_wait_asynccnt(8);
  else     __builtin_amdgcn_s_wait_asynccnt(0);
#else
  if (tag) asm volatile("s_wait_asynccnt 0x8" ::: "memory");
  else     asm volatile("s_wait_asynccnt 0x0" ::: "memory");
#endif
#endif
  (void)tag;
}

union Frag16 { v16bf v; uint4 q[2]; };

// A fragment, 16x32 bf16 (ISA 7.12.2): lane L holds row M=L%16;
// K base = 8*(L/16); VGPR0-3 = K kb..kb+7, VGPR4-7 = K kb+16..kb+23.
__device__ __forceinline__ v16bf load_frag_a(const __bf16* s, int lane, int rbase, int ks) {
  const int r  = rbase + (lane & 15);
  const int kb = ks * 32 + ((lane >> 4) << 3);
  const __bf16* p = s + r * LDT + kb;
  Frag16 f;
  f.q[0] = *(const uint4*)(p);
  f.q[1] = *(const uint4*)(p + 16);
  return f.v;
}

// B fragment, 32x16 bf16: lane L holds column N=L%16; K base = 8*(L/16);
// VGPR0-3 = K kb..kb+7, VGPR4-7 = K kb+16..kb+23 (matches A fragment).
__device__ __forceinline__ v16bf load_frag_b(const __bf16* s, int lane, int cbase, int ks) {
  const int c  = cbase + (lane & 15);
  const int kb = ks * 32 + ((lane >> 4) << 3);
  const __bf16* p = s + c * LDT + kb;
  Frag16 f;
  f.q[0] = *(const uint4*)(p);
  f.q[1] = *(const uint4*)(p + 16);
  return f.v;
}

// MODE 0: out bf16                      (pre -> h)
// MODE 1: out bf16, relu                (g1 -> t1, e1 -> x1, e2 -> x2)
// MODE 2: dyn/rew weighted accumulation (eo -> dynAcc/rewAcc)
// MODE 3: out f32 + residual aux0 (z)   (head next_z)
// MODE 4: out f32, col-guarded          (head reward, N=101)
// B is pre-transposed n-major: Bt[n][k], row stride ldB(=K), rows padded with
// zeros up to the 128-multiple, so staging has no guards at all.
template <int MODE>
__global__ __launch_bounds__(256, 1)
void gemm_bf16_wmma(const __bf16* __restrict__ A, const __bf16* __restrict__ Bt,
                    const float* __restrict__ bias,
                    float* __restrict__ outF, __bf16* __restrict__ outB,
                    const float* __restrict__ aux0, const float* __restrict__ aux1,
                    float* __restrict__ acc0, float* __restrict__ acc1,
                    int M, int N, int K, int ldA, int ldB, int ldOut,
                    int expert, int initAcc)
{
  __shared__ __bf16 sA[2][TM * LDT];
  __shared__ __bf16 sB[2][TN * LDT];

  const int tid  = threadIdx.x;
  const int lane = tid & 31;
  const int wave = tid >> 5;
  const int wr   = wave >> 1;   // 0..3 -> 32-row slab
  const int wc   = wave & 1;    // 0..1 -> 64-col slab

  const int m0 = blockIdx.y * TM;
  const int n0 = blockIdx.x * TN;

  v8f acc[2][4];
  #pragma unroll
  for (int i = 0; i < 2; ++i)
    #pragma unroll
    for (int j = 0; j < 4; ++j)
      acc[i][j] = v8f{};

  // both tiles: 128 rows x 64 k bf16; thread t stages row t/2, k-half t%2
  const int srow = tid >> 1;          // 0..127
  const int scol = (tid & 1) * 32;    // 0 / 32 (bf16)

  auto stage = [&](int k0, int buf) {
    const __bf16* ga = A  + (size_t)(m0 + srow) * ldA + (k0 + scol);
    const __bf16* gb = Bt + (size_t)(n0 + srow) * ldB + (k0 + scol);
    __bf16* la = sA[buf] + srow * LDT + scol;
    __bf16* lb = sB[buf] + srow * LDT + scol;
    #pragma unroll
    for (int i = 0; i < 4; ++i) {
#if defined(USE_ASYNC_LDS)
      __builtin_amdgcn_global_load_async_to_lds_b128(
          (AS1 v4i*)(ga + i * 8), (AS3 v4i*)(la + i * 8), 0, 0);
      __builtin_amdgcn_global_load_async_to_lds_b128(
          (AS1 v4i*)(gb + i * 8), (AS3 v4i*)(lb + i * 8), 0, 0);
#else
      *(uint4*)(la + i * 8) = *(const uint4*)(ga + i * 8);
      *(uint4*)(lb + i * 8) = *(const uint4*)(gb + i * 8);
#endif
    }
  };

  auto compute = [&](int buf) {
    #pragma unroll
    for (int ks = 0; ks < 2; ++ks) {
      v16bf afr[2], bfr[4];
      #pragma unroll
      for (int i = 0; i < 2; ++i)
        afr[i] = load_frag_a(sA[buf], lane, wr * 32 + i * 16, ks);
      #pragma unroll
      for (int j = 0; j < 4; ++j)
        bfr[j] = load_frag_b(sB[buf], lane, wc * 64 + j * 16, ks);
      #pragma unroll
      for (int i = 0; i < 2; ++i)
        #pragma unroll
        for (int j = 0; j < 4; ++j)
          acc[i][j] = __builtin_amdgcn_wmma_f32_16x16x32_bf16(
              false, afr[i], false, bfr[j], (short)0, acc[i][j], false, false);
    }
  };

  stage(0, 0);                         // prologue: 8 asyncs in flight
  int cur = 0;
  for (int k0 = 0; k0 < K; k0 += TK) {
    const bool haveNext = (k0 + TK) < K;
    if (haveNext) stage(k0 + TK, cur ^ 1);  // 8 more asyncs -> other buffer
    wait_async_le(haveNext ? 1 : 0);        // older stage's 8 retired
    __syncthreads();
    compute(cur);
    cur ^= 1;
  }

  // ---- epilogue (C/D layout: lane<16 -> M=v, lane>=16 -> M=8+v; N=lane%16) --
  const int rB = m0 + wr * 32;
  const int cB = n0 + wc * 64;
  #pragma unroll
  for (int i = 0; i < 2; ++i) {
    #pragma unroll
    for (int j = 0; j < 4; ++j) {
      #pragma unroll
      for (int v = 0; v < 8; ++v) {
        const int r = rB + i * 16 + ((lane >> 4) << 3) + v;
        const int c = cB + j * 16 + (lane & 15);
        if (c < N) {
          float val = acc[i][j][v] + bias[c];
          if (MODE == 0) {
            outB[(size_t)r * ldOut + c] = (__bf16)val;
          } else if (MODE == 1) {
            outB[(size_t)r * ldOut + c] = (__bf16)fmaxf(val, 0.0f);
          } else if (MODE == 2) {
            const float wd = aux0[(size_t)r * 8 + expert];
            const float wv = aux1[(size_t)r * 8 + expert];
            const size_t idx = (size_t)r * ldOut + c;
            if (initAcc) { acc0[idx] = wd * val;  acc1[idx] = wv * val; }
            else         { acc0[idx] += wd * val; acc1[idx] += wv * val; }
          } else if (MODE == 3) {
            outF[(size_t)r * ldOut + c] = val + aux0[(size_t)r * ldOut + c];
          } else {
            outF[(size_t)r * ldOut + c] = val;
          }
        }
      }
    }
  }
}

// ---------------------------------------------------------------------------
// gating GEMM (K=1024 -> 16 logits) + dual softmax, one wave per row
// ---------------------------------------------------------------------------
__global__ __launch_bounds__(256)
void g2_softmax(const __bf16* __restrict__ t1, const float* __restrict__ W,
                const float* __restrict__ b, float* __restrict__ wdyn,
                float* __restrict__ wrew, int rows)
{
  const int gw   = (int)((blockIdx.x * blockDim.x + threadIdx.x) >> 5);
  const int lane = threadIdx.x & 31;
  if (gw >= rows) return;

  const __bf16* tr = t1 + (size_t)gw * 1024;
  float acc[16];
  #pragma unroll
  for (int j = 0; j < 16; ++j) acc[j] = 0.0f;

  for (int k = lane; k < 1024; k += 32) {
    const float tv = (float)tr[k];
    const float* w = W + (size_t)k * 16;
    #pragma unroll
    for (int j = 0; j < 16; ++j) acc[j] = fmaf(tv, w[j], acc[j]);
  }
  #pragma unroll
  for (int j = 0; j < 16; ++j) {
    #pragma unroll
    for (int off = 16; off > 0; off >>= 1)
      acc[j] += __shfl_down(acc[j], off, 32);
  }
  if (lane == 0) {
    float lg[16];
    #pragma unroll
    for (int j = 0; j < 16; ++j) lg[j] = acc[j] + b[j];
    float m0 = lg[0], m1 = lg[8];
    #pragma unroll
    for (int j = 1; j < 8; ++j) { m0 = fmaxf(m0, lg[j]); m1 = fmaxf(m1, lg[8 + j]); }
    float e0[8], e1[8], s0 = 0.0f, s1 = 0.0f;
    #pragma unroll
    for (int j = 0; j < 8; ++j) {
      e0[j] = __expf(lg[j] - m0);     s0 += e0[j];
      e1[j] = __expf(lg[8 + j] - m1); s1 += e1[j];
    }
    const float r0 = 1.0f / s0, r1 = 1.0f / s1;
    #pragma unroll
    for (int j = 0; j < 8; ++j) {
      wdyn[(size_t)gw * 8 + j] = e0[j] * r0;
      wrew[(size_t)gw * 8 + j] = e1[j] * r1;
    }
  }
}

// ---------------------------------------------------------------------------
// helpers
// ---------------------------------------------------------------------------
// W (K x ldW f32, using Ncols columns) -> Wt (Np x K bf16, n-major, zero pad)
__global__ __launch_bounds__(256)
void transpose_to_bf16(const float* __restrict__ W, __bf16* __restrict__ Wt,
                       int K, int ldW, int Ncols, int Np)
{
  const size_t total = (size_t)Np * K;
  for (size_t i = blockIdx.x * (size_t)blockDim.x + threadIdx.x; i < total;
       i += (size_t)gridDim.x * blockDim.x) {
    const int n = (int)(i / K), k = (int)(i % K);
    const float v = (n < Ncols) ? W[(size_t)k * ldW + n] : 0.0f;
    Wt[i] = (__bf16)v;
  }
}

__global__ __launch_bounds__(256)
void f32_to_bf16(const float* __restrict__ in, __bf16* __restrict__ out, size_t n)
{
  for (size_t i = blockIdx.x * (size_t)blockDim.x + threadIdx.x; i < n;
       i += (size_t)gridDim.x * blockDim.x)
    out[i] = (__bf16)in[i];
}

__global__ __launch_bounds__(256)
void concat_bf16(const float* __restrict__ z, const float* __restrict__ a,
                 __bf16* __restrict__ out, size_t n)
{
  for (size_t i = blockIdx.x * (size_t)blockDim.x + threadIdx.x; i < n;
       i += (size_t)gridDim.x * blockDim.x) {
    const size_t row = i / 576, c = i % 576;
    const float v = (c < 512) ? z[row * 512 + c] : a[row * 64 + (c - 512)];
    out[i] = (__bf16)v;
  }
}

// ---------------------------------------------------------------------------
extern "C" void kernel_launch(void* const* d_in, const int* in_sizes, int n_in,
                              void* d_out, int out_size, void* d_ws, size_t ws_size,
                              hipStream_t stream)
{
  constexpr int NB = 16384, LAT = 512, HID = 1024, E = 8, RD = 101, RDP = 128;

  const float* z    = (const float*)d_in[0];
  const float* av   = (const float*)d_in[1];
  const float* preW = (const float*)d_in[2];
  const float* preB = (const float*)d_in[3];
  const float* g1W  = (const float*)d_in[4];
  const float* g1B  = (const float*)d_in[5];
  const float* g2W  = (const float*)d_in[6];
  const float* g2B  = (const float*)d_in[7];
  const float* e1W  = (const float*)d_in[8];
  const float* e1B  = (const float*)d_in[9];
  const float* e2W  = (const float*)d_in[10];
  const float* e2B  = (const float*)d_in[11];
  const float* eoW  = (const float*)d_in[12];
  const float* eoB  = (const float*)d_in[13];
  const float* hW   = (const float*)d_in[14];
  const float* hBb  = (const float*)d_in[15];

  char* ws = (char*)d_ws;
  size_t off = 0;
  auto take = [&](size_t elems, size_t esz) -> void* {
    void* p = ws + off;
    off += (elems * esz + 255) & ~(size_t)255;
    return p;
  };
  __bf16* catB   = (__bf16*)take((size_t)NB * 576, 2);
  __bf16* hAct   = (__bf16*)take((size_t)NB * HID, 2);
  __bf16* bufA   = (__bf16*)take((size_t)NB * HID, 2);
  __bf16* bufB   = (__bf16*)take((size_t)NB * HID, 2);
  float*  dynAcc = (float*) take((size_t)NB * LAT, 4);
  float*  rewAcc = (float*) take((size_t)NB * LAT, 4);
  __bf16* preWt  = (__bf16*)take((size_t)HID * 576, 2);       // 1024 x 576
  __bf16* g1Wt   = (__bf16*)take((size_t)HID * HID, 2);       // 1024 x 1024
  __bf16* e1Wt   = (__bf16*)take((size_t)E * HID * HID, 2);   // 8 x 1024 x 1024
  __bf16* e2Wt   = (__bf16*)take((size_t)E * HID * HID, 2);
  __bf16* eoWt   = (__bf16*)take((size_t)E * LAT * HID, 2);   // 8 x 512 x 1024
  __bf16* h0t    = (__bf16*)take((size_t)LAT * LAT, 2);       // 512 x 512
  __bf16* h1t    = (__bf16*)take((size_t)RDP * LAT, 2);       // 128 x 512 (pad)
  __bf16* dynB   = catB;   // catB dead after pre-GEMM
  __bf16* rewB   = bufA;   // bufA dead after expert loop

  float* out    = (float*)d_out;
  float* o_next = out;
  float* o_rew  = out + (size_t)NB * LAT;
  float* o_wd   = o_rew + (size_t)NB * RD;
  float* o_wr   = o_wd + (size_t)NB * E;

  const int CV = 2048;
  // weight f32 -> transposed/padded bf16 (bandwidth-trivial vs ~760 GF GEMMs)
  transpose_to_bf16<<<CV, 256, 0, stream>>>(preW, preWt, 576, HID, HID, HID);
  transpose_to_bf16<<<CV, 256, 0, stream>>>(g1W,  g1Wt,  HID, HID, HID, HID);
  for (int e = 0; e < E; ++e) {
    transpose_to_bf16<<<CV, 256, 0, stream>>>(
        e1W + (size_t)e * HID * HID, e1Wt + (size_t)e * HID * HID, HID, HID, HID, HID);
    transpose_to_bf16<<<CV, 256, 0, stream>>>(
        e2W + (size_t)e * HID * HID, e2Wt + (size_t)e * HID * HID, HID, HID, HID, HID);
    transpose_to_bf16<<<CV, 256, 0, stream>>>(
        eoW + (size_t)e * HID * LAT, eoWt + (size_t)e * LAT * HID, HID, LAT, LAT, LAT);
  }
  transpose_to_bf16<<<CV, 256, 0, stream>>>(hW, h0t, LAT, LAT, LAT, LAT);
  transpose_to_bf16<<<CV, 256, 0, stream>>>(hW + (size_t)LAT * LAT, h1t, LAT, LAT, RD, RDP);
  concat_bf16<<<CV, 256, 0, stream>>>(z, av, catB, (size_t)NB * 576);

  const dim3 blk(256);
  auto grid = [](int N_, int M_) { return dim3((N_ + TN - 1) / TN, M_ / TM); };

  // pre: h = cat @ preW + b          (16384x576x1024)
  gemm_bf16_wmma<0><<<grid(HID, NB), blk, 0, stream>>>(
      catB, preWt, preB, nullptr, hAct, nullptr, nullptr, nullptr, nullptr,
      NB, HID, 576, 576, 576, HID, 0, 0);
  // g1: t1 = relu(h @ g1W + b)       (16384x1024x1024)
  gemm_bf16_wmma<1><<<grid(HID, NB), blk, 0, stream>>>(
      hAct, g1Wt, g1B, nullptr, bufA, nullptr, nullptr, nullptr, nullptr,
      NB, HID, HID, HID, HID, HID, 0, 0);
  // gating + softmax -> w_dyn, w_rew (written straight into d_out)
  g2_softmax<<<NB / 8, 256, 0, stream>>>(bufA, g2W, g2B, o_wd, o_wr, NB);

  for (int e = 0; e < E; ++e) {
    // x1 = relu(h @ e1W[e] + b)
    gemm_bf16_wmma<1><<<grid(HID, NB), blk, 0, stream>>>(
        hAct, e1Wt + (size_t)e * HID * HID, e1B + (size_t)e * HID,
        nullptr, bufA, nullptr, nullptr, nullptr, nullptr,
        NB, HID, HID, HID, HID, HID, 0, 0);
    // x2 = relu(x1 @ e2W[e] + b)
    gemm_bf16_wmma<1><<<grid(HID, NB), blk, 0, stream>>>(
        bufA, e2Wt + (size_t)e * HID * HID, e2B + (size_t)e * HID,
        nullptr, bufB, nullptr, nullptr, nullptr, nullptr,
        NB, HID, HID, HID, HID, HID, 0, 0);
    // feats_e = x2 @ eoW[e] + b ; dynAcc += w_dyn[:,e]*feats ; rewAcc likewise
    gemm_bf16_wmma<2><<<grid(LAT, NB), blk, 0, stream>>>(
        bufB, eoWt + (size_t)e * LAT * HID, eoB + (size_t)e * LAT,
        nullptr, nullptr, o_wd, o_wr, dynAcc, rewAcc,
        NB, LAT, HID, HID, HID, LAT, e, (e == 0) ? 1 : 0);
  }

  f32_to_bf16<<<CV, 256, 0, stream>>>(dynAcc, dynB, (size_t)NB * LAT);
  f32_to_bf16<<<CV, 256, 0, stream>>>(rewAcc, rewB, (size_t)NB * LAT);

  // next_z = dyn @ headW0 + b0 + z
  gemm_bf16_wmma<3><<<grid(LAT, NB), blk, 0, stream>>>(
      dynB, h0t, hBb, o_next, nullptr, z, nullptr, nullptr, nullptr,
      NB, LAT, LAT, LAT, LAT, LAT, 0, 0);
  // reward = rew @ headW1[:, :101] + b1[:101]
  gemm_bf16_wmma<4><<<grid(RD, NB), blk, 0, stream>>>(
      rewB, h1t, hBb + LAT, o_rew, nullptr,
      nullptr, nullptr, nullptr, nullptr,
      NB, RD, LAT, LAT, LAT, RD, 0, 0);
}
